// MPNN_1Conv_50818053046295
// MI455X (gfx1250) — matrix operand order repaired and checked by reading, not compiled, not run on hardware
//
#include <hip/hip_runtime.h>

// MI455X / gfx1250: wave32, WMMA bf16 16x16x32 with f32 accumulate.
// Reference collapses: node features are zeroed, so conv layer0 is
// edge_attr @ conv_w0[64:96,:] + b0.  Whole edge path fused in one kernel,
// scatter-mean via f32 atomics in L2, node path in a second kernel.

typedef __bf16 bf16_t;
typedef __attribute__((ext_vector_type(16))) __bf16 v16bf;
typedef __attribute__((ext_vector_type(8)))  float  v8f;

#define CH 32

__device__ __forceinline__ float leaky_f(float x) { return x > 0.f ? x : 0.2f * x; }

__device__ __forceinline__ v8f wmma_bf16(v16bf a, v16bf b, v8f c) {
  // (neg_a, A, neg_b, B, c_mod, C, reuse_a, reuse_b)
  return __builtin_amdgcn_wmma_f32_16x16x32_bf16(false, a, false, b, (short)0, c, false, false);
}

__device__ __forceinline__ v8f splat8(float x) {
  v8f c;
#pragma unroll
  for (int r = 0; r < 8; ++r) c[r] = x;
  return c;
}

// B layout for V_WMMA_F32_16X16X32_BF16 (B is 32x16):
// lane n = lane&15 selects column N; lanes 0-15 hold K in {0..7, 16..23}, lanes 16-31 K in {8..15, 24..31},
// element i within a lane = K listed in order.  w is a 32x32 row-major bf16 weight
// (in_features x out_features), t selects the N half.
__device__ __forceinline__ v16bf load_B(const bf16_t* w, int lane, int t) {
  const int n  = (lane & 15) + 16 * t;
  const int kb = (lane >> 4) * 8;
  v16bf b;
#pragma unroll
  for (int i = 0; i < 8; ++i) {
    b[i]     = w[(kb + i) * CH + n];
    b[8 + i] = w[(16 + kb + i) * CH + n];
  }
  return b;
}

// D/C layout: lane l holds column N=(l&15)+16t; element r holds row M=r+8*(l>>4).
// Store both N-halves as a 16x32 bf16 row-major tile in (wave-private) LDS,
// applying activation: 0=none, 1=relu, 2=leaky.
template <int ACT>
__device__ __forceinline__ void stash(bf16_t* act, v8f c0, v8f c1, int lane) {
  const int n  = lane & 15;
  const int mb = (lane >> 4) * 8;
#pragma unroll
  for (int r = 0; r < 8; ++r) {
    float x0 = c0[r], x1 = c1[r];
    if (ACT == 1) { x0 = fmaxf(x0, 0.f); x1 = fmaxf(x1, 0.f); }
    if (ACT == 2) { x0 = leaky_f(x0);    x1 = leaky_f(x1); }
    act[(mb + r) * CH + n]      = (bf16_t)x0;
    act[(mb + r) * CH + n + 16] = (bf16_t)x1;
  }
}

// A layout for 16-bit A 16x32: lane m=l&15 is the row; lanes 0-15 hold
// K in {0..7, 16..23}, lanes 16-31 hold K in {8..15, 24..31}.
__device__ __forceinline__ v16bf loadA_lds(const bf16_t* act, int lane) {
  const int m  = lane & 15;
  const int cb = (lane >> 4) * 8;
  v16bf a;
#pragma unroll
  for (int i = 0; i < 8; ++i) {
    a[i]     = act[m * CH + cb + i];
    a[8 + i] = act[m * CH + 16 + cb + i];
  }
  return a;
}

__global__ void __launch_bounds__(256) zero_ws_kernel(float* p, long n) {
  long i = (long)blockIdx.x * blockDim.x + threadIdx.x;
  long stride = (long)gridDim.x * blockDim.x;
  for (; i < n; i += stride) p[i] = 0.f;
}

__global__ void __launch_bounds__(256) mpnn_edge_kernel(
    const float* __restrict__ edge_attr, const int* __restrict__ edge_index,
    const float* __restrict__ cw0, const float* __restrict__ cb0,
    const float* __restrict__ cw1, const float* __restrict__ cb1,
    const float* __restrict__ ew0, const float* __restrict__ eb0,
    const float* __restrict__ ew1, const float* __restrict__ eb1,
    const float* __restrict__ ew2, const float* __restrict__ eb2,
    float* __restrict__ out_edge, float* __restrict__ sums, float* __restrict__ cnt,
    int E, int numTiles)
{
  __shared__ bf16_t wl[5][CH][CH];       // bf16 weights: conv0(rows64..95), conv1, emlp0..2
  __shared__ bf16_t act[8][16 * CH];     // per-wave transpose scratch

  // Cooperative weight staging, f32 -> bf16 (conv_w0 only rows 64..95 matter).
  bf16_t* wf = &wl[0][0][0];
  for (int i = threadIdx.x; i < CH * CH; i += 256) {
    wf[0 * 1024 + i] = (bf16_t)cw0[64 * CH + i];
    wf[1 * 1024 + i] = (bf16_t)cw1[i];
    wf[2 * 1024 + i] = (bf16_t)ew0[i];
    wf[3 * 1024 + i] = (bf16_t)ew1[i];
    wf[4 * 1024 + i] = (bf16_t)ew2[i];
  }
  __syncthreads();

  const int lane = threadIdx.x & 31;
  const int wid  = threadIdx.x >> 5;
  bf16_t* myact  = act[wid];

  // B-matrix registers (5 layers x 2 N-halves) and per-lane biases.
  v16bf B00 = load_B(&wl[0][0][0], lane, 0), B01 = load_B(&wl[0][0][0], lane, 1);
  v16bf B10 = load_B(&wl[1][0][0], lane, 0), B11 = load_B(&wl[1][0][0], lane, 1);
  v16bf B20 = load_B(&wl[2][0][0], lane, 0), B21 = load_B(&wl[2][0][0], lane, 1);
  v16bf B30 = load_B(&wl[3][0][0], lane, 0), B31 = load_B(&wl[3][0][0], lane, 1);
  v16bf B40 = load_B(&wl[4][0][0], lane, 0), B41 = load_B(&wl[4][0][0], lane, 1);
  const int n = lane & 15;
  float b0l = cb0[n], b0h = cb0[n + 16];
  float b1l = cb1[n], b1h = cb1[n + 16];
  float b2l = eb0[n], b2h = eb0[n + 16];
  float b3l = eb1[n], b3h = eb1[n + 16];
  float b4l = eb2[n], b4h = eb2[n + 16];

  const int* rowIdx = edge_index;       // edge_index[0] = destination rows
  const int m  = lane & 15;
  const int cb = (lane >> 4) * 8;
  const int mb = (lane >> 4) * 8;

  const int gw = blockIdx.x * 8 + wid;
  const int nw = gridDim.x * 8;
  for (int tile = gw; tile < numTiles; tile += nw) {
    const int base = tile * 16;

    // ---- A tile from global edge_attr (f32 -> bf16) ----
    int e0 = base + m; if (e0 > E - 1) e0 = E - 1;
    const float* src = edge_attr + (size_t)e0 * CH;
    v16bf a;
#pragma unroll
    for (int i = 0; i < 8; ++i) {
      a[i]     = (bf16_t)src[cb + i];
      a[8 + i] = (bf16_t)src[16 + cb + i];
    }

    // ---- conv layer 0 + relu ----
    v8f c0 = splat8(b0l), c1 = splat8(b0h);
    c0 = wmma_bf16(a, B00, c0);
    c1 = wmma_bf16(a, B01, c1);
    stash<1>(myact, c0, c1, lane);          // relu
    a = loadA_lds(myact, lane);             // DS in-order within wave: no barrier

    // ---- conv layer 1 -> edge_out (no activation) ----
    c0 = splat8(b1l); c1 = splat8(b1h);
    c0 = wmma_bf16(a, B10, c0);
    c1 = wmma_bf16(a, B11, c1);

    // ---- scatter-add edge_out into node sums (L2-resident, f32 atomics) ----
#pragma unroll
    for (int r = 0; r < 8; ++r) {
      int e = base + mb + r;
      if (e < E) {
        int v = rowIdx[e];
        atomicAdd(&sums[(size_t)v * CH + n],      c0[r]);
        atomicAdd(&sums[(size_t)v * CH + n + 16], c1[r]);
      }
    }
    if (lane < 16) {
      int e = base + lane;
      if (e < E) atomicAdd(&cnt[rowIdx[e]], 1.0f);
    }

    // ---- edge_1 = leaky(edge_out); edge MLP (leaky, leaky, none) ----
    stash<2>(myact, c0, c1, lane);
    a = loadA_lds(myact, lane);

    c0 = splat8(b2l); c1 = splat8(b2h);
    c0 = wmma_bf16(a, B20, c0);
    c1 = wmma_bf16(a, B21, c1);
    stash<2>(myact, c0, c1, lane);
    a = loadA_lds(myact, lane);

    c0 = splat8(b3l); c1 = splat8(b3h);
    c0 = wmma_bf16(a, B30, c0);
    c1 = wmma_bf16(a, B31, c1);
    stash<2>(myact, c0, c1, lane);
    a = loadA_lds(myact, lane);

    c0 = splat8(b4l); c1 = splat8(b4h);
    c0 = wmma_bf16(a, B40, c0);
    c1 = wmma_bf16(a, B41, c1);

    // ---- write edge output ----
#pragma unroll
    for (int r = 0; r < 8; ++r) {
      int e = base + mb + r;
      if (e < E) {
        out_edge[(size_t)e * CH + n]      = c0[r];
        out_edge[(size_t)e * CH + n + 16] = c1[r];
      }
    }
  }
}

__global__ void __launch_bounds__(256) mpnn_node_kernel(
    const float* __restrict__ sums, const float* __restrict__ cnt,
    const float* __restrict__ nw0, const float* __restrict__ nb0,
    const float* __restrict__ nw1, const float* __restrict__ nb1,
    const float* __restrict__ nw2, const float* __restrict__ nb2,
    float* __restrict__ out_node, int N, int numTiles)
{
  __shared__ bf16_t wl[3][CH][CH];
  __shared__ bf16_t act[8][16 * CH];

  bf16_t* wf = &wl[0][0][0];
  for (int i = threadIdx.x; i < CH * CH; i += 256) {
    wf[0 * 1024 + i] = (bf16_t)nw0[i];
    wf[1 * 1024 + i] = (bf16_t)nw1[i];
    wf[2 * 1024 + i] = (bf16_t)nw2[i];
  }
  __syncthreads();

  const int lane = threadIdx.x & 31;
  const int wid  = threadIdx.x >> 5;
  bf16_t* myact  = act[wid];

  v16bf B00 = load_B(&wl[0][0][0], lane, 0), B01 = load_B(&wl[0][0][0], lane, 1);
  v16bf B10 = load_B(&wl[1][0][0], lane, 0), B11 = load_B(&wl[1][0][0], lane, 1);
  v16bf B20 = load_B(&wl[2][0][0], lane, 0), B21 = load_B(&wl[2][0][0], lane, 1);
  const int n = lane & 15;
  float b0l = nb0[n], b0h = nb0[n + 16];
  float b1l = nb1[n], b1h = nb1[n + 16];
  float b2l = nb2[n], b2h = nb2[n + 16];

  const int m  = lane & 15;
  const int cb = (lane >> 4) * 8;
  const int mb = (lane >> 4) * 8;

  const int gw = blockIdx.x * 8 + wid;
  const int nw = gridDim.x * 8;
  for (int tile = gw; tile < numTiles; tile += nw) {
    const int base = tile * 16;

    // ---- A tile = leaky(sums / max(cnt,1)) ----
    int v0 = base + m; if (v0 > N - 1) v0 = N - 1;
    const float inv = 1.0f / fmaxf(cnt[v0], 1.0f);
    const float* src = sums + (size_t)v0 * CH;
    v16bf a;
#pragma unroll
    for (int i = 0; i < 8; ++i) {
      a[i]     = (bf16_t)leaky_f(src[cb + i] * inv);
      a[8 + i] = (bf16_t)leaky_f(src[16 + cb + i] * inv);
    }

    v8f c0 = splat8(b0l), c1 = splat8(b0h);
    c0 = wmma_bf16(a, B00, c0);
    c1 = wmma_bf16(a, B01, c1);
    stash<2>(myact, c0, c1, lane);
    a = loadA_lds(myact, lane);

    c0 = splat8(b1l); c1 = splat8(b1h);
    c0 = wmma_bf16(a, B10, c0);
    c1 = wmma_bf16(a, B11, c1);
    stash<2>(myact, c0, c1, lane);
    a = loadA_lds(myact, lane);

    c0 = splat8(b2l); c1 = splat8(b2h);
    c0 = wmma_bf16(a, B20, c0);
    c1 = wmma_bf16(a, B21, c1);

#pragma unroll
    for (int r = 0; r < 8; ++r) {
      int v = base + mb + r;
      if (v < N) {
        out_node[(size_t)v * CH + n]      = c0[r];
        out_node[(size_t)v * CH + n + 16] = c1[r];
      }
    }
  }
}

extern "C" void kernel_launch(void* const* d_in, const int* in_sizes, int n_in,
                              void* d_out, int out_size, void* d_ws, size_t ws_size,
                              hipStream_t stream) {
  // setup_inputs() order: node_attr, edge_attr, edge_index, then params in
  // insertion order: conv_w0, conv_b0, conv_w1, conv_b1,
  //                  node_mlp_{w0,b0,w1,b1,w2,b2}, edge_mlp_{w0,b0,w1,b1,w2,b2}
  const float* edge_attr  = (const float*)d_in[1];
  const int*   edge_index = (const int*)d_in[2];
  const float* cw0 = (const float*)d_in[3];
  const float* cb0 = (const float*)d_in[4];
  const float* cw1 = (const float*)d_in[5];
  const float* cb1 = (const float*)d_in[6];
  const float* nw0 = (const float*)d_in[7];
  const float* nb0 = (const float*)d_in[8];
  const float* nw1 = (const float*)d_in[9];
  const float* nb1 = (const float*)d_in[10];
  const float* nw2 = (const float*)d_in[11];
  const float* nb2 = (const float*)d_in[12];
  const float* ew0 = (const float*)d_in[13];
  const float* eb0 = (const float*)d_in[14];
  const float* ew1 = (const float*)d_in[15];
  const float* eb1 = (const float*)d_in[16];
  const float* ew2 = (const float*)d_in[17];
  const float* eb2 = (const float*)d_in[18];

  const int N = in_sizes[0] / CH;
  const int E = in_sizes[1] / CH;

  float* sums = (float*)d_ws;                 // N x 32
  float* cnt  = sums + (size_t)N * CH;        // N
  float* out_node = (float*)d_out;            // N x 32 (first tuple element)
  float* out_edge = out_node + (size_t)N * CH;// E x 32 (second tuple element)

  const long zcount = (long)N * CH + N;
  zero_ws_kernel<<<512, 256, 0, stream>>>(sums, zcount);

  const int tilesE = (E + 15) / 16;
  mpnn_edge_kernel<<<1024, 256, 0, stream>>>(
      edge_attr, edge_index, cw0, cb0, cw1, cb1,
      ew0, eb0, ew1, eb1, ew2, eb2,
      out_edge, sums, cnt, E, tilesE);

  const int tilesN = (N + 15) / 16;
  mpnn_node_kernel<<<128, 256, 0, stream>>>(
      sums, cnt, nw0, nb0, nw1, nb1, nw2, nb2,
      out_node, N, tilesN);
}
